// SelfAttention_38431367365424
// MI455X (gfx1250) — compile-verified, matrix operand order checked by reading, not run on hardware
//
#include <hip/hip_runtime.h>
#include <hip/hip_bf16.h>

typedef __attribute__((ext_vector_type(16))) _Float16 v16h;
typedef __attribute__((ext_vector_type(8)))  _Float16 v8h;
typedef __attribute__((ext_vector_type(8)))  float    v8f;

// ---- constants for this problem ----
#define BATCH 8
#define CCH   256
#define NPIX  1024
#define HEADS 8
#define HD    32
#define NGRP  8

static __device__ __forceinline__ v8f wmma_f16(v16h a, v16h b, v8f c) {
  // v_wmma_f32_16x16x32_f16: (neg_a, A, neg_b, B, c_mod, C, reuse_a, reuse_b)
  return __builtin_amdgcn_wmma_f32_16x16x32_f16(false, a, false, b, (short)0, c, false, false);
}

// Build a 16-half fragment from two contiguous 8-half (16B) chunks.
// A-layout: lane l<16 holds row l, K {k0..k0+7} then {k0+16..k0+23} with k0=(l>>4)*8.
// B-layout mirrored with columns.
static __device__ __forceinline__ v16h make_frag(const _Float16* p0, const _Float16* p1) {
  v8h lo = *(const v8h*)p0;
  v8h hi = *(const v8h*)p1;
  return __builtin_shufflevector(lo, hi, 0,1,2,3,4,5,6,7,8,9,10,11,12,13,14,15);
}

// ---------------- kernel 1: weight f32 -> f16 ----------------
__global__ void k_convert(const float* __restrict__ wq, const float* __restrict__ wo,
                          _Float16* __restrict__ wq16, _Float16* __restrict__ wo16) {
  int i = blockIdx.x * 256 + threadIdx.x;
  if (i < 3 * CCH * CCH) wq16[i] = (_Float16)wq[i];
  if (i < CCH * CCH)     wo16[i] = (_Float16)wo[i];
}

// ---------------- kernel 2: GroupNorm -> h_t [B, N, C] f16 ----------------
__global__ void k_groupnorm(const float* __restrict__ x, const float* __restrict__ gs,
                            const float* __restrict__ gb, _Float16* __restrict__ h16) {
  __shared__ float s_sum[256];
  __shared__ float s_sq[256];
  int b = blockIdx.x >> 3, g = blockIdx.x & 7;
  int tid = threadIdx.x;
  const float* xb = x + (size_t)(b * CCH + g * 32) * NPIX;  // 32 channels x 1024 pixels
  float s = 0.f, ss = 0.f;
  for (int i = tid; i < 32 * NPIX; i += 256) { float v = xb[i]; s += v; ss += v * v; }
  s_sum[tid] = s; s_sq[tid] = ss;
  __syncthreads();
  for (int o = 128; o > 0; o >>= 1) {
    if (tid < o) { s_sum[tid] += s_sum[tid + o]; s_sq[tid] += s_sq[tid + o]; }
    __syncthreads();
  }
  float mean = s_sum[0] * (1.f / 32768.f);
  float var  = s_sq[0] * (1.f / 32768.f) - mean * mean;
  float inv  = rsqrtf(var + 1e-5f);
  for (int i = tid; i < 32 * NPIX; i += 256) {
    int cl = i >> 10, n = i & 1023;
    int c = g * 32 + cl;
    float v = (xb[i] - mean) * inv * gs[c] + gb[c];
    h16[((size_t)(b * NPIX + n)) * CCH + c] = (_Float16)v;
  }
}

// ---------------- kernel 3: QKV GEMM (WMMA) ----------------
// qkv[o,n] = sum_c w[o,c] * h[c,n]; A = w_qkv_f16 [768,256] row-major,
// B = h_t [B][n][c] (col n of B is a contiguous 256-half row).
__global__ void k_qkv(const _Float16* __restrict__ wq16, const _Float16* __restrict__ h16,
                      const float* __restrict__ bqkv,
                      _Float16* __restrict__ q16, _Float16* __restrict__ k16,
                      _Float16* __restrict__ v16) {
  int wave = (blockIdx.x * blockDim.x + threadIdx.x) >> 5;  // 8*48*64 waves
  int lane = threadIdx.x & 31;
  int b = wave / (48 * 64);
  int rem = wave % (48 * 64);
  int mt = rem / 64, nt = rem % 64;
  int r0 = lane & 15, kh = (lane >> 4) * 8;

  const _Float16* arow = wq16 + (size_t)(mt * 16 + r0) * CCH + kh;
  const _Float16* brow = h16 + ((size_t)(b * NPIX + nt * 16 + r0)) * CCH + kh;
  v8f acc = {};
#pragma unroll
  for (int kc = 0; kc < CCH; kc += 32) {
    v16h a  = make_frag(arow + kc, arow + kc + 16);
    v16h bm = make_frag(brow + kc, brow + kc + 16);
    acc = wmma_f16(a, bm, acc);
  }
  int n = nt * 16 + r0;
#pragma unroll
  for (int r = 0; r < 8; ++r) {
    int o = mt * 16 + r + 8 * (lane >> 4);
    float val = acc[r] + bqkv[o];
    int sec = o >> 8;            // 0:q 1:k 2:v
    int oc = o & 255;
    int head = oc >> 5, d = oc & 31;
    size_t bh = (size_t)(b * HEADS + head);
    if (sec == 0)       q16[(bh * NPIX + n) * HD + d] = (_Float16)(val * 0.17677669529663687f);
    else if (sec == 1)  k16[(bh * NPIX + n) * HD + d] = (_Float16)val;
    else                v16[(bh * HD + d) * NPIX + n] = (_Float16)val;
  }
}

// ---------------- kernel 4: fused flash attention ----------------
// block = (b, head, n-tile of 128). 8 waves; wave owns 16 rows.
// LDS: sQ[128][32] + sK[1024][32] + sV[32][1024] + sP[8][16][32] (f16) = 144KB
__global__ void k_attn(const _Float16* __restrict__ q16, const _Float16* __restrict__ k16,
                       const _Float16* __restrict__ v16, _Float16* __restrict__ a16) {
  extern __shared__ _Float16 smem[];
  _Float16* sQ = smem;                 // 128*32
  _Float16* sK = sQ + 128 * 32;        // 1024*32
  _Float16* sV = sK + 1024 * 32;       // 32*1024
  _Float16* sP = sV + 32 * 1024;       // 8*16*32

  int blk = blockIdx.x;                // 512 blocks
  int b = blk >> 6;
  int rem = blk & 63;
  int head = rem >> 3, ntile = rem & 7;
  int tid = threadIdx.x, lane = tid & 31, w = tid >> 5;
  size_t bh = (size_t)(b * HEADS + head);

  const _Float16* qg = q16 + (bh * NPIX + ntile * 128) * HD;
  const _Float16* kg = k16 + bh * NPIX * HD;
  const _Float16* vg = v16 + bh * HD * NPIX;
  for (int i = tid; i < 128 * 32 / 8; i += 256) ((v8h*)sQ)[i] = ((const v8h*)qg)[i];
  for (int i = tid; i < 1024 * 32 / 8; i += 256) ((v8h*)sK)[i] = ((const v8h*)kg)[i];
  for (int i = tid; i < 32 * 1024 / 8; i += 256) ((v8h*)sV)[i] = ((const v8h*)vg)[i];
  __syncthreads();

  int r0 = lane & 15, hb = lane >> 4, kh = hb * 8;
  const _Float16* qrow = sQ + (w * 16 + r0) * 32 + kh;
  v16h aq = make_frag(qrow, qrow + 16);

  v8f o0 = {}, o1 = {};
  float rmax[8], rsum[8];
#pragma unroll
  for (int r = 0; r < 8; ++r) { rmax[r] = -1e30f; rsum[r] = 0.f; }
  _Float16* sPw = sP + w * 16 * 32;

  for (int mc = 0; mc < NPIX; mc += 32) {
    const _Float16* kc0 = sK + (mc + r0) * 32 + kh;
    const _Float16* kc1 = sK + (mc + 16 + r0) * 32 + kh;
    v16h bk0 = make_frag(kc0, kc0 + 16);
    v16h bk1 = make_frag(kc1, kc1 + 16);
    v8f z = {};
    v8f s0 = wmma_f16(aq, bk0, z);   // rows: r+8*(l/16), cols: mc + l%16
    v8f s1 = wmma_f16(aq, bk1, z);   // cols: mc+16 + l%16

#pragma unroll
    for (int r = 0; r < 8; ++r) {
      float t = fmaxf(s0[r], s1[r]);
      t = fmaxf(t, __shfl_xor(t, 1, 32));
      t = fmaxf(t, __shfl_xor(t, 2, 32));
      t = fmaxf(t, __shfl_xor(t, 4, 32));
      t = fmaxf(t, __shfl_xor(t, 8, 32));
      float nm = fmaxf(rmax[r], t);
      float fac = __expf(rmax[r] - nm);
      rmax[r] = nm;
      float e0 = __expf(s0[r] - nm);
      float e1 = __expf(s1[r] - nm);
      float ps = e0 + e1;
      ps += __shfl_xor(ps, 1, 32);
      ps += __shfl_xor(ps, 2, 32);
      ps += __shfl_xor(ps, 4, 32);
      ps += __shfl_xor(ps, 8, 32);
      rsum[r] = rsum[r] * fac + ps;
      o0[r] *= fac;
      o1[r] *= fac;
      int rr = r + 8 * hb;
      sPw[rr * 32 + r0]      = (_Float16)e0;   // P[row][m-local]
      sPw[rr * 32 + 16 + r0] = (_Float16)e1;
    }
    // cross-lane LDS transpose of P: wait for per-wave stores to land
    asm volatile("s_wait_dscnt 0" ::: "memory");

    const _Float16* prow = sPw + r0 * 32 + kh;
    v16h ap = make_frag(prow, prow + 16);                        // A: P [16n x 32m]
    const _Float16* vc0 = sV + r0 * NPIX + mc + kh;              // B cols d=0..15
    const _Float16* vc1 = sV + (16 + r0) * NPIX + mc + kh;       // B cols d=16..31
    v16h bv0 = make_frag(vc0, vc0 + 16);
    v16h bv1 = make_frag(vc1, vc1 + 16);
    o0 = wmma_f16(ap, bv0, o0);
    o1 = wmma_f16(ap, bv1, o1);
  }

#pragma unroll
  for (int r = 0; r < 8; ++r) {
    float inv = 1.f / rsum[r];
    int n = ntile * 128 + w * 16 + r + 8 * hb;
    _Float16* dst = a16 + ((size_t)(b * NPIX + n)) * CCH + head * HD;
    dst[r0]      = (_Float16)(o0[r] * inv);
    dst[16 + r0] = (_Float16)(o1[r] * inv);
  }
}

// ---------------- kernel 5: out projection + bias + residual ----------------
__global__ void k_proj(const _Float16* __restrict__ wo16, const _Float16* __restrict__ a16,
                       const float* __restrict__ bo, const float* __restrict__ x,
                       float* __restrict__ out) {
  int wave = (blockIdx.x * blockDim.x + threadIdx.x) >> 5;  // 8*16*64 waves
  int lane = threadIdx.x & 31;
  int b = wave / (16 * 64);
  int rem = wave % (16 * 64);
  int mt = rem / 64, nt = rem % 64;
  int r0 = lane & 15, kh = (lane >> 4) * 8;

  const _Float16* arow = wo16 + (size_t)(mt * 16 + r0) * CCH + kh;
  const _Float16* brow = a16 + ((size_t)(b * NPIX + nt * 16 + r0)) * CCH + kh;
  v8f acc = {};
#pragma unroll
  for (int kc = 0; kc < CCH; kc += 32) {
    v16h a  = make_frag(arow + kc, arow + kc + 16);
    v16h bm = make_frag(brow + kc, brow + kc + 16);
    acc = wmma_f16(a, bm, acc);
  }
  int n = nt * 16 + r0;
#pragma unroll
  for (int r = 0; r < 8; ++r) {
    int o = mt * 16 + r + 8 * (lane >> 4);
    size_t idx = ((size_t)(b * CCH + o)) * NPIX + n;
    out[idx] = x[idx] + bo[o] + acc[r];
  }
}

extern "C" void kernel_launch(void* const* d_in, const int* in_sizes, int n_in,
                              void* d_out, int out_size, void* d_ws, size_t ws_size,
                              hipStream_t stream) {
  const float* x     = (const float*)d_in[0];
  const float* gscal = (const float*)d_in[1];
  const float* gbias = (const float*)d_in[2];
  const float* wqkv  = (const float*)d_in[3];
  const float* bqkv  = (const float*)d_in[4];
  const float* wout  = (const float*)d_in[5];
  const float* bout  = (const float*)d_in[6];
  float* out = (float*)d_out;

  char* ws = (char*)d_ws;
  _Float16* wq16 = (_Float16*)(ws);                        // 768*256     -> 393216 B
  _Float16* wo16 = (_Float16*)(ws + 393216);               // 256*256     -> 131072 B
  _Float16* h16  = (_Float16*)(ws + 524288);               // 8*1024*256  -> 4 MiB
  _Float16* q16  = (_Float16*)(ws + 4718592);              // [bh][n][d]  -> 4 MiB
  _Float16* k16  = (_Float16*)(ws + 8912896);              // [bh][n][d]  -> 4 MiB
  _Float16* v16  = (_Float16*)(ws + 13107200);             // [bh][d][n]  -> 4 MiB
  _Float16* a16  = (_Float16*)(ws + 17301504);             // [b][n][c]   -> 4 MiB

  k_convert<<<768, 256, 0, stream>>>(wqkv, wout, wq16, wo16);
  k_groupnorm<<<BATCH * NGRP, 256, 0, stream>>>(x, gscal, gbias, h16);
  k_qkv<<<(BATCH * 48 * 64) / 4, 128, 0, stream>>>(wq16, h16, bqkv, q16, k16, v16);
  size_t attn_lds = (size_t)(128 * 32 + 1024 * 32 + 32 * 1024 + 8 * 16 * 32) * sizeof(_Float16);
  k_attn<<<BATCH * HEADS * (NPIX / 128), 256, attn_lds, stream>>>(q16, k16, v16, a16);
  k_proj<<<(BATCH * 16 * 64) / 4, 128, 0, stream>>>(wo16, a16, bout, x, out);
}
